// Conv2d_NN_spatial_20435454394615
// MI455X (gfx1250) — matrix operand order repaired and checked by reading, not compiled, not run on hardware
//
#include <hip/hip_runtime.h>

typedef _Float16 half16 __attribute__((ext_vector_type(16)));
typedef _Float16 half8  __attribute__((ext_vector_type(8)));
typedef float    float8 __attribute__((ext_vector_type(8)));

union H16 { half16 v; half8 h[2]; };

#define B_   16
#define C_   16
#define H_   224
#define W_   224
#define hS_  112
#define N_   12544      // 112*112 tokens
#define M_   256        // sampled tokens
#define C1_  72
#define CP_  96         // C1 padded to 3*32 for K-loop
#define KD_  216        // C1*K
#define KDP_ 224        // padded to 7*32
#define NT_  784        // N_/16 token tiles
#define MT_  5          // 80/16 output-channel tiles

// ---- helpers -------------------------------------------------------------

__device__ __forceinline__ int sample_coord(int g) {
  // round(linspace(0,111,16))[g] ; g*7.4 never lands on .5 so trunc(x+0.5) == round
  return (int)((float)g * 7.4f + 0.5f);
}

// value of unshuffled channel c1 at token (i,j): c1 = CH*4 + s1*2 + s2
__device__ __forceinline__ float gather_c1(const float* x, int b, int c1, int i, int j) {
  int CH = c1 >> 2, s1 = (c1 >> 1) & 1, s2 = c1 & 1;
  int y = 2 * i + s1, xx = 2 * j + s2;
  if (CH < C_) return x[(((size_t)b * C_ + CH) * H_ + y) * W_ + xx];
  float fy = (float)y, fx = (float)xx;
  float d = fmaxf(sqrtf(fy * fy + fx * fx), 1e-12f);
  return (CH == C_) ? (fy / d) : (fx / d);
}

__device__ __forceinline__ float swz_f(float v) {  // lane ^ 16 exchange (SWAPX16)
  return __builtin_bit_cast(float, __builtin_amdgcn_ds_swizzle(__builtin_bit_cast(int, v), 0x401F));
}
__device__ __forceinline__ int swz_i(int v) {
  return __builtin_amdgcn_ds_swizzle(v, 0x401F);
}

// stable comparator matching jax.lax.top_k tie-breaking (lower index wins ties)
__device__ __forceinline__ bool better(float xv, int xi, float yv, int yi) {
  return (xv > yv) || (xv == yv && xi < yi);
}

// ---- 1) normalized token matrix: (B, N, 96) f16 --------------------------

__global__ void build_tokens(const float* __restrict__ x, _Float16* __restrict__ tok) {
  int t = blockIdx.x * blockDim.x + threadIdx.x;   // exactly B_*N_ threads
  int b = t / N_, n = t % N_;
  int i = n / hS_, j = n % hS_;
  float vals[C1_];
  float ss = 0.f;
#pragma unroll
  for (int c1 = 0; c1 < C1_; ++c1) {
    float v = gather_c1(x, b, c1, i, j);
    vals[c1] = v;
    ss += v * v;
  }
  float inv = 1.0f / fmaxf(sqrtf(ss), 1e-12f);
  _Float16* o = tok + (size_t)t * CP_;
#pragma unroll
  for (int g = 0; g < CP_ / 8; ++g) {              // 12 x b128 stores
    half8 p;
#pragma unroll
    for (int e = 0; e < 8; ++e) {
      int c1 = g * 8 + e;
      p[e] = (c1 < C1_) ? (_Float16)(vals[c1] * inv) : (_Float16)0.f;
    }
    *(half8*)(o + g * 8) = p;
  }
}

// ---- 2) sample matrices: norm (B,256,96) f16, raw (B,256,72) f16 ---------

__global__ void build_samples(const float* __restrict__ x,
                              _Float16* __restrict__ snorm,
                              _Float16* __restrict__ sraw) {
  int t = blockIdx.x * blockDim.x + threadIdx.x;   // exactly B_*M_ threads
  int b = t / M_, m = t % M_;
  int i = sample_coord(m >> 4), j = sample_coord(m & 15);
  float vals[C1_];
  float ss = 0.f;
#pragma unroll
  for (int c1 = 0; c1 < C1_; ++c1) {
    float v = gather_c1(x, b, c1, i, j);
    vals[c1] = v;
    ss += v * v;
  }
  float inv = 1.0f / fmaxf(sqrtf(ss), 1e-12f);
  _Float16* on = snorm + (size_t)t * CP_;
  _Float16* orw = sraw + (size_t)t * C1_;
#pragma unroll
  for (int g = 0; g < CP_ / 8; ++g) {
    half8 p;
#pragma unroll
    for (int e = 0; e < 8; ++e) {
      int c1 = g * 8 + e;
      p[e] = (c1 < C1_) ? (_Float16)(vals[c1] * inv) : (_Float16)0.f;
    }
    *(half8*)(on + g * 8) = p;
  }
#pragma unroll
  for (int g = 0; g < C1_ / 8; ++g) {              // 9 x b128 raw stores
    half8 p;
#pragma unroll
    for (int e = 0; e < 8; ++e) p[e] = (_Float16)vals[g * 8 + e];
    *(half8*)(orw + g * 8) = p;
  }
}

// ---- 3) pack w1 (72,72,3) f32 -> (80,224) f16, zero padded ----------------

__global__ void pack_w1(const float* __restrict__ w1, _Float16* __restrict__ w1p) {
  int t = blockIdx.x * blockDim.x + threadIdx.x;
  if (t >= 80 * KDP_) return;
  int o = t / KDP_, q = t % KDP_;          // q = c*3 + k matches w1 [o][c][k] flat
  float v = (o < C1_ && q < KD_) ? w1[o * KD_ + q] : 0.f;
  w1p[t] = (_Float16)v;
}

// ---- 4) fused cosine-sim GEMM + top-3 (WMMA f16 -> f32) -------------------
// A = samples (M-axis), B = this wave's 16 tokens (N-axis). Token fragments
// are per-wave loop-invariant -> hoisted (loaded once). Top-3 kept in
// registers, branch-free updates; lanes l and l^16 merge via ds_swizzle.

__global__ void sim_topk(const _Float16* __restrict__ tok,
                         const _Float16* __restrict__ snorm,
                         int* __restrict__ top3) {
  const int lane = threadIdx.x & 31;
  const int wave = blockIdx.x * (blockDim.x >> 5) + (threadIdx.x >> 5); // [0, B_*NT_)
  const int b = wave / NT_;
  const int ntile = wave % NT_;
  const int col = lane & 15;
  const int hi = lane >> 4;
  const int token = ntile * 16 + col;

  const _Float16* tokb = tok + ((size_t)b * N_ + token) * CP_;
  const _Float16* sampb = snorm + (size_t)b * M_ * CP_;

  // hoist the 3 token (B-operand) fragments: loaded once per wave
  H16 bb[3];
#pragma unroll
  for (int kt = 0; kt < 3; ++kt) {
    const _Float16* bp = tokb + kt * 32 + hi * 8;
    bb[kt].h[0] = *(const half8*)(bp);
    bb[kt].h[1] = *(const half8*)(bp + 16);
  }

  float v0 = -1e30f, v1 = -1e30f, v2 = -1e30f;
  int i0 = 0, i1 = 0, i2 = 0;

  for (int mt = 0; mt < 16; ++mt) {
    float8 c = {0.f, 0.f, 0.f, 0.f, 0.f, 0.f, 0.f, 0.f};
#pragma unroll
    for (int kt = 0; kt < 3; ++kt) {
      H16 a;
      const _Float16* ap = sampb + (size_t)(mt * 16 + col) * CP_ + kt * 32 + hi * 8;
      a.h[0] = *(const half8*)(ap);
      a.h[1] = *(const half8*)(ap + 16);
      c = __builtin_amdgcn_wmma_f32_16x16x32_f16(false, a.v, false, bb[kt].v,
                                                 (short)0, c, false, false);
    }
#pragma unroll
    for (int v = 0; v < 8; ++v) {
      float s = c[v];
      int sid = mt * 16 + hi * 8 + v;          // ascending scan -> stable ties
      bool g0 = s > v0, g1 = s > v1, g2 = s > v2;
      float n2 = g1 ? v1 : (g2 ? s : v2);   int j2 = g1 ? i1 : (g2 ? sid : i2);
      float n1 = g0 ? v0 : (g1 ? s : v1);   int j1 = g0 ? i0 : (g1 ? sid : i1);
      float n0 = g0 ? s  : v0;              int j0 = g0 ? sid : i0;
      v0 = n0; v1 = n1; v2 = n2; i0 = j0; i1 = j1; i2 = j2;
    }
  }

  // merge with partner lane (same token, other 8-row half of each tile)
  float bv0 = swz_f(v0), bv1 = swz_f(v1), bv2 = swz_f(v2);
  int   bi0 = swz_i(i0), bi1 = swz_i(i1), bi2 = swz_i(i2);

  bool  c0   = better(v0, i0, bv0, bi0);
  int   r0i  = c0 ? i0 : bi0;
  float ca1v = c0 ? v1 : v0;  int ca1i = c0 ? i1 : i0;
  float cb1v = c0 ? bv0 : bv1; int cb1i = c0 ? bi0 : bi1;
  bool  c1   = better(ca1v, ca1i, cb1v, cb1i);
  int   r1i  = c1 ? ca1i : cb1i;
  float ca2v = c1 ? (c0 ? v2 : v1) : ca1v;  int ca2i = c1 ? (c0 ? i2 : i1) : ca1i;
  float cb2v = c1 ? cb1v : (c0 ? bv1 : bv2); int cb2i = c1 ? cb1i : (c0 ? bi1 : bi2);
  bool  c2   = better(ca2v, ca2i, cb2v, cb2i);
  int   r2i  = c2 ? ca2i : cb2i;

  if (hi == 0) {
    int* o = top3 + ((size_t)b * N_ + token) * 3;
    o[0] = r0i; o[1] = r1i; o[2] = r2i;
  }
}

// ---- 5) gather neighbors into P: (B, N, 224) f16 --------------------------
// Reads only the 256-sample raw table (L2 resident, ~36 KB/batch); writes b128.

__global__ void gather_pack(const _Float16* __restrict__ sraw,
                            const int* __restrict__ top3,
                            _Float16* __restrict__ P) {
  unsigned t = blockIdx.x * blockDim.x + threadIdx.x;  // exactly B_*N_*28 threads
  int qg = t % (KDP_ / 8);
  unsigned bn = t / (KDP_ / 8);              // b*N_ + n
  int b = bn / N_;
  int s0 = top3[(size_t)bn * 3 + 0];
  int s1 = top3[(size_t)bn * 3 + 1];
  int s2 = top3[(size_t)bn * 3 + 2];
  const _Float16* base = sraw + (size_t)b * M_ * C1_;
  half8 p;
#pragma unroll
  for (int e = 0; e < 8; ++e) {
    int q = qg * 8 + e;
    _Float16 v = (_Float16)0.f;
    if (q < KD_) {
      int c = q / 3, k = q - c * 3;
      int sid = (k == 0) ? s0 : ((k == 1) ? s1 : s2);
      v = base[sid * C1_ + c];
    }
    p[e] = v;
  }
  *(half8*)(P + (size_t)bn * KDP_ + qg * 8) = p;
}

// ---- 6) conv1d as GEMM (80x224)@(224xN) with WMMA, +bias -> x3 (B,72,N) ---
// One wave owns a 16-token column strip and ALL 5 output-channel tiles:
// each P fragment is loaded once per kt and reused by 5 WMMAs (5 accumulators
// = 40 VGPRs). w1 fragments are 35 KB total -> L0/L2 resident.

__global__ void conv_gemm(const _Float16* __restrict__ P,
                          const _Float16* __restrict__ w1p,
                          const float* __restrict__ b1,
                          float* __restrict__ x3) {
  const int lane = threadIdx.x & 31;
  const int wave = blockIdx.x * (blockDim.x >> 5) + (threadIdx.x >> 5); // [0, B_*NT_)
  const int b = wave / NT_;
  const int ntile = wave % NT_;
  const int col = lane & 15;
  const int hi = lane >> 4;

  const _Float16* pb = P + ((size_t)b * N_ + ntile * 16 + col) * KDP_;

  float8 c[MT_];
#pragma unroll
  for (int mt = 0; mt < MT_; ++mt)
    c[mt] = (float8){0.f, 0.f, 0.f, 0.f, 0.f, 0.f, 0.f, 0.f};

#pragma unroll
  for (int kt = 0; kt < 7; ++kt) {
    H16 bbf;                                  // P column fragment: load once
    const _Float16* bp = pb + kt * 32 + hi * 8;
    bbf.h[0] = *(const half8*)(bp);
    bbf.h[1] = *(const half8*)(bp + 16);
#pragma unroll
    for (int mt = 0; mt < MT_; ++mt) {
      H16 a;
      const _Float16* ap = w1p + (size_t)(mt * 16 + col) * KDP_ + kt * 32 + hi * 8;
      a.h[0] = *(const half8*)(ap);
      a.h[1] = *(const half8*)(ap + 16);
      c[mt] = __builtin_amdgcn_wmma_f32_16x16x32_f16(false, a.v, false, bbf.v,
                                                     (short)0, c[mt], false, false);
    }
  }
  const int token = ntile * 16 + col;
#pragma unroll
  for (int mt = 0; mt < MT_; ++mt) {
#pragma unroll
    for (int v = 0; v < 8; ++v) {
      int o = mt * 16 + v + hi * 8;
      if (o < C1_)
        x3[((size_t)b * C1_ + o) * N_ + token] = c[mt][v] + b1[o];
    }
  }
}

// ---- 7) pixel-shuffle + pointwise conv (18 -> 16) -------------------------

__global__ void shuffle_pw(const float* __restrict__ x3,
                           const float* __restrict__ w_pw,
                           const float* __restrict__ b_pw,
                           float* __restrict__ out) {
  int t = blockIdx.x * blockDim.x + threadIdx.x;   // exactly B_*H_*W_ threads
  int xx = t % W_;
  int rest = t / W_;
  int y = rest % H_;
  int b = rest / H_;
  int s = (y & 1) * 2 + (xx & 1);
  int n = (y >> 1) * hS_ + (xx >> 1);
  float vals[18];
#pragma unroll
  for (int CH = 0; CH < 18; ++CH)
    vals[CH] = x3[((size_t)b * C1_ + CH * 4 + s) * N_ + n];
#pragma unroll
  for (int o = 0; o < C_; ++o) {
    float acc = b_pw[o];
#pragma unroll
    for (int CH = 0; CH < 18; ++CH) acc += w_pw[o * 18 + CH] * vals[CH];
    out[(((size_t)b * C_ + o) * H_ + y) * W_ + xx] = acc;
  }
}

// ---- host launch ----------------------------------------------------------

extern "C" void kernel_launch(void* const* d_in, const int* in_sizes, int n_in,
                              void* d_out, int out_size, void* d_ws, size_t ws_size,
                              hipStream_t stream) {
  const float* x    = (const float*)d_in[0];
  const float* w1   = (const float*)d_in[1];
  const float* b1   = (const float*)d_in[2];
  const float* w_pw = (const float*)d_in[3];
  const float* b_pw = (const float*)d_in[4];
  float* out = (float*)d_out;

  char* ws = (char*)d_ws;
  size_t off = 0;
  auto alloc = [&](size_t bytes) -> void* {
    void* p = ws + off;
    off = (off + bytes + 255) & ~(size_t)255;
    return p;
  };

  _Float16* tok   = (_Float16*)alloc((size_t)B_ * N_ * CP_ * 2);   // 38.5 MB
  _Float16* snorm = (_Float16*)alloc((size_t)B_ * M_ * CP_ * 2);   // 0.8 MB
  _Float16* sraw  = (_Float16*)alloc((size_t)B_ * M_ * C1_ * 2);   // 0.6 MB
  int*      top3  = (int*)alloc((size_t)B_ * N_ * 3 * 4);          // 2.4 MB
  _Float16* w1p   = (_Float16*)alloc((size_t)80 * KDP_ * 2);       // 36 KB
  _Float16* P     = (_Float16*)alloc((size_t)B_ * N_ * KDP_ * 2);  // 89.9 MB
  float*    x3    = (float*)alloc((size_t)B_ * C1_ * N_ * 4);      // 57.8 MB
  (void)ws_size; (void)in_sizes; (void)n_in; (void)out_size;

  const int TPB = 256;

  // 1) tokens: B_*N_ = 200704 threads -> 784 blocks (exact)
  build_tokens<<<(B_ * N_) / TPB, TPB, 0, stream>>>(x, tok);
  // 2) samples: 4096 threads -> 16 blocks (exact)
  build_samples<<<(B_ * M_) / TPB, TPB, 0, stream>>>(x, snorm, sraw);
  // 3) w1 pack: 17920 threads
  pack_w1<<<(80 * KDP_ + TPB - 1) / TPB, TPB, 0, stream>>>(w1, w1p);
  // 4) sim + top3: B_*NT_ = 12544 waves, 8 waves/block -> 1568 blocks (exact)
  sim_topk<<<(B_ * NT_) / 8, TPB, 0, stream>>>(tok, snorm, top3);
  // 5) gather P: B_*N_*28 = 5.62M threads -> 21952 blocks (exact)
  gather_pack<<<(B_ * N_ * (KDP_ / 8)) / TPB, TPB, 0, stream>>>(sraw, top3, P);
  // 6) conv GEMM: B_*NT_ = 12544 waves -> 1568 blocks (exact)
  conv_gemm<<<(B_ * NT_) / 8, TPB, 0, stream>>>(P, w1p, b1, x3);
  // 7) shuffle + pointwise: B_*H_*W_ = 802816 threads -> 3136 blocks (exact)
  shuffle_pw<<<(B_ * H_ * W_) / TPB, TPB, 0, stream>>>(x3, w_pw, b_pw, out);
}
